// E3Relax_40192303956691
// MI455X (gfx1250) — compile-verified, matrix operand order checked by reading, not run on hardware
//
#include <hip/hip_runtime.h>

typedef __attribute__((ext_vector_type(16))) _Float16 v16h;
typedef __attribute__((ext_vector_type(8)))  float    v8f;

namespace {
constexpr int   HD    = 128;
constexpr int   NRBF  = 128;
constexpr int   NNODE = 10000;
constexpr int   NGR   = 8;
constexpr int   NEDGE = 200000;
constexpr int   EPG   = NEDGE / NGR;
constexpr int   NPG   = NNODE / NGR;
constexpr float CUT   = 6.0f;
constexpr float SSC   = 1.0f / 0.6f;
constexpr float IS3   = 0.57735026918962576f;
constexpr float ISH   = 0.08838834764831845f;   // 1/sqrt(128)
constexpr float IS2   = 0.70710678118654752f;
constexpr float RBFC  = -8064.5f;               // -0.5 * 127^2
}

// ---------------- device helpers ----------------

__device__ __forceinline__ void atomAddF(float* p, float v) {
  __hip_atomic_fetch_add(p, v, __ATOMIC_RELAXED, __HIP_MEMORY_SCOPE_AGENT);
}

__device__ __forceinline__ float ssilu(float v) {
  return v / (1.0f + __expf(-v)) * SSC;
}

__device__ __forceinline__ float envelope(float ds) {
  if (ds >= 1.0f) return 0.0f;
  float d2 = ds * ds;
  float d5 = d2 * d2 * ds;
  return 1.0f + (-21.0f) * d5 + 35.0f * d5 * ds + (-15.0f) * d5 * d2;
}

union HFrag { v16h v; uint4 q[2]; _Float16 h[16]; };

__device__ __forceinline__ v8f do_wmma(v16h a, v16h b, v8f c) {
  return __builtin_amdgcn_wmma_f32_16x16x32_f16(false, a, false, b, (short)0, c, false, false);
}

// ---------------- weight transpose+convert: W[K,N] f32 -> Wt[N,K] f16 --------
__global__ void k_cvtT(const float* __restrict__ W, _Float16* __restrict__ Wt,
                       int K, int N) {
  int i = blockIdx.x * blockDim.x + threadIdx.x;
  if (i >= K * N) return;
  int k = i / N, n = i % N;
  Wt[n * K + k] = (_Float16)W[i];
}

// ---------------- generic WMMA GEMM ----------------
// Y[M,N] = act(X[M,K] @ W[K,N] + bias), W given transposed f16 as Wt[N,K].
// 256 threads = 8 waves; block tile 32(M) x 128(N); each wave: 16x32 (2 WMMAs/K-step).
__global__ void k_gemm(const float* __restrict__ X, const _Float16* __restrict__ Wt,
                       const float* __restrict__ bias, float* __restrict__ Y,
                       int M, int N, int K, int act) {
  __shared__ __align__(16) _Float16 As[32 * 32];
  int wave = threadIdx.x >> 5;
  int lane = threadIdx.x & 31;
  int m0 = blockIdx.y * 32 + (wave >> 2) * 16;
  int n0 = blockIdx.x * 128 + (wave & 3) * 32;
  int arow = (wave >> 2) * 16;
  v8f c0 = {}, c1 = {};
  for (int k0 = 0; k0 < K; k0 += 32) {
    __syncthreads();
    {
      int r  = threadIdx.x >> 3;          // 0..31
      int c4 = (threadIdx.x & 7) * 4;     // 0,4,..,28
      int row = blockIdx.y * 32 + r;
      float4 v = {0.0f, 0.0f, 0.0f, 0.0f};
      if (row < M) v = *(const float4*)(X + (size_t)row * K + k0 + c4);
      union { _Float16 h[4]; uint2 u; } pk;
      pk.h[0] = (_Float16)v.x; pk.h[1] = (_Float16)v.y;
      pk.h[2] = (_Float16)v.z; pk.h[3] = (_Float16)v.w;
      *(uint2*)&As[r * 32 + c4] = pk.u;
    }
    __syncthreads();
    if (n0 < N) {
      int kb = (lane & 16) ? 8 : 0;
      int rr = arow + (lane & 15);
      HFrag a, b0, b1;
      a.q[0] = *(const uint4*)&As[rr * 32 + kb];
      a.q[1] = *(const uint4*)&As[rr * 32 + kb + 16];
      const _Float16* p0 = Wt + (size_t)(n0 + (lane & 15)) * K + k0 + kb;
      const _Float16* p1 = p0 + (size_t)16 * K;
      if (k0 + 32 < K) __builtin_prefetch(p0 + 32, 0, 0);
      b0.q[0] = *(const uint4*)p0;  b0.q[1] = *(const uint4*)(p0 + 16);
      b1.q[0] = *(const uint4*)p1;  b1.q[1] = *(const uint4*)(p1 + 16);
      c0 = do_wmma(a.v, b0.v, c0);
      c1 = do_wmma(a.v, b1.v, c1);
    }
  }
  if (n0 >= N) return;
  int rbase = (lane & 16) ? 8 : 0;
#pragma unroll
  for (int tt = 0; tt < 2; ++tt) {
    int col  = n0 + tt * 16 + (lane & 15);
    float bv = bias ? bias[col] : 0.0f;
    const v8f& c = tt ? c1 : c0;
#pragma unroll
    for (int r = 0; r < 8; ++r) {
      int row = m0 + rbase + r;
      if (row < M) {
        float v = c[r] + bv;
        if (act == 1) v = ssilu(v);
        Y[row * N + col] = v;
      }
    }
  }
}

// ---------------- fused edge message passing ----------------
// 16 edges / block (256 thr = 8 waves). RBF -> LDS f16, WMMA vs edge_proj Wt
// (3 accumulators per wave, A fragment reused), gather x_h[j], vec[j],
// scatter-add into d_x / d_vec with f32 atomics.
__global__ void k_edge_mp(const float* __restrict__ pos, const float* __restrict__ cell,
                          const float* __restrict__ cof, const int* __restrict__ jidx,
                          const int* __restrict__ iidx, const float* __restrict__ xh,
                          const float* __restrict__ vec, const _Float16* __restrict__ WepT,
                          const float* __restrict__ Bep, float* __restrict__ dx,
                          float* __restrict__ dvec) {
  __shared__ __align__(16) _Float16 Rf[16 * 128];
  __shared__ float Sud[16 * 3];
  __shared__ int Sj[16], Si[16];

  int e0   = blockIdx.x * 16;
  int le   = threadIdx.x >> 4;   // local edge 0..15
  int part = threadIdx.x & 15;   // 8 RBF values each
  int e    = e0 + le;
  int ej = jidx[e], ei = iidx[e];
  int b  = e / EPG;
  float pd[3];
#pragma unroll
  for (int v = 0; v < 3; ++v) {
    float s = pos[ej * 3 + v] - pos[ei * 3 + v];
#pragma unroll
    for (int p = 0; p < 3; ++p) s += cof[e * 3 + p] * cell[b * 9 + p * 3 + v];
    pd[v] = s;
  }
  float dist = sqrtf(pd[0] * pd[0] + pd[1] * pd[1] + pd[2] * pd[2]);
  if (part == 0) {
    Sj[le] = ej; Si[le] = ei;
#pragma unroll
    for (int v = 0; v < 3; ++v) Sud[le * 3 + v] = -pd[v] / dist;
  }
  float ds  = dist / CUT;
  float env = envelope(ds);
  {
    union { _Float16 h[8]; uint4 q; } pk;
#pragma unroll
    for (int t = 0; t < 8; ++t) {
      int r   = part * 8 + t;
      float d = ds - (float)r * (1.0f / 127.0f);
      pk.h[t] = (_Float16)(env * __expf(RBFC * d * d));
    }
    *(uint4*)&Rf[le * 128 + part * 8] = pk.q;
  }
  __syncthreads();

  int wave = threadIdx.x >> 5;
  int lane = threadIdx.x & 31;
  int cols[3];
#pragma unroll
  for (int t = 0; t < 3; ++t) cols[t] = (wave * 3 + t) * 16 + (lane & 15);
  v8f acc0 = {}, acc1 = {}, acc2 = {};
  for (int k0 = 0; k0 < 128; k0 += 32) {
    int kb = k0 + ((lane & 16) ? 8 : 0);
    int rr = lane & 15;
    HFrag a;
    a.q[0] = *(const uint4*)&Rf[rr * 128 + kb];
    a.q[1] = *(const uint4*)&Rf[rr * 128 + kb + 16];
    HFrag b0, b1, b2;
    const _Float16* p0 = WepT + (size_t)cols[0] * 128 + kb;
    const _Float16* p1 = WepT + (size_t)cols[1] * 128 + kb;
    const _Float16* p2 = WepT + (size_t)cols[2] * 128 + kb;
    b0.q[0] = *(const uint4*)p0;  b0.q[1] = *(const uint4*)(p0 + 16);
    b1.q[0] = *(const uint4*)p1;  b1.q[1] = *(const uint4*)(p1 + 16);
    b2.q[0] = *(const uint4*)p2;  b2.q[1] = *(const uint4*)(p2 + 16);
    acc0 = do_wmma(a.v, b0.v, acc0);
    acc1 = do_wmma(a.v, b1.v, acc1);
    acc2 = do_wmma(a.v, b2.v, acc2);
  }
#pragma unroll
  for (int t = 0; t < 3; ++t) {
    const v8f& c = (t == 0) ? acc0 : (t == 1) ? acc1 : acc2;
    int col   = cols[t];
    int third = col >> 7;
    int hh    = col & 127;
    float bv  = Bep[col];
    int rbase = (lane & 16) ? 8 : 0;
#pragma unroll
    for (int r = 0; r < 8; ++r) {
      int m  = rbase + r;
      int jj = Sj[m], ii = Si[m];
      float val = (c[r] + bv) * xh[jj * 384 + col] * IS3;
      if (third == 0) {
#pragma unroll
        for (int d = 0; d < 3; ++d)
          atomAddF(&dvec[(ii * 3 + d) * HD + hh], val * vec[(jj * 3 + d) * HD + hh] * ISH);
      } else if (third == 1) {
#pragma unroll
        for (int d = 0; d < 3; ++d)
          atomAddF(&dvec[(ii * 3 + d) * HD + hh], val * Sud[m * 3 + d] * ISH);
      } else {
        atomAddF(&dx[ii * HD + hh], val);
      }
    }
  }
}

// ---------------- elementwise / small kernels ----------------

__global__ void k_zero(float* __restrict__ p, int n) {
  int i = blockIdx.x * blockDim.x + threadIdx.x;
  if (i < n) p[i] = 0.0f;
}
__global__ void k_copy(const float* __restrict__ s, float* __restrict__ d, int n) {
  int i = blockIdx.x * blockDim.x + threadIdx.x;
  if (i < n) d[i] = s[i];
}
__global__ void k_add(float* __restrict__ y, const float* __restrict__ a, int n) {
  int i = blockIdx.x * blockDim.x + threadIdx.x;
  if (i < n) y[i] += a[i];
}
__global__ void k_embed(const float* __restrict__ emb, const int* __restrict__ z,
                        float* __restrict__ x) {
  int i = blockIdx.x * blockDim.x + threadIdx.x;
  if (i < NNODE * HD) x[i] = emb[z[i / HD] * HD + (i % HD)];
}
__global__ void k_init_scal(const float* __restrict__ semb, float* __restrict__ scal) {
  int i = blockIdx.x * blockDim.x + threadIdx.x;
  if (i >= 3 * NGR * HD) return;
  int k = i / (NGR * HD), h = i % HD;
  scal[i] = semb[k * HD + h];
}
__global__ void k_node_rbf(const float* __restrict__ pos, const float* __restrict__ cell,
                           const int* __restrict__ batch, int k, float* __restrict__ rbf) {
  int i = blockIdx.x * blockDim.x + threadIdx.x;
  if (i >= NNODE * NRBF) return;
  int n = i / NRBF, r = i % NRBF;
  int b = batch[n];
  float dx = cell[b * 9 + k * 3 + 0] - pos[n * 3 + 0];
  float dy = cell[b * 9 + k * 3 + 1] - pos[n * 3 + 1];
  float dz = cell[b * 9 + k * 3 + 2] - pos[n * 3 + 2];
  float dist = sqrtf(dx * dx + dy * dy + dz * dz);
  float ds = dist / CUT;
  float d  = ds - (float)r * (1.0f / 127.0f);
  rbf[i] = envelope(ds) * __expf(RBFC * d * d);
}
__global__ void k_latloc_apply(float* __restrict__ x, float* __restrict__ vec,
                               const float* __restrict__ ep, const float* __restrict__ xp,
                               const float* __restrict__ pos, const float* __restrict__ cell,
                               const int* __restrict__ batch, int k) {
  int i = blockIdx.x * blockDim.x + threadIdx.x;
  if (i >= NNODE * HD) return;
  int n = i / HD, h = i % HD;
  float g1 = ep[n * 384 + h]       * xp[n * 384 + h]       * IS3;
  float g2 = ep[n * 384 + 128 + h] * xp[n * 384 + 128 + h] * IS3;
  float g3 = ep[n * 384 + 256 + h] * xp[n * 384 + 256 + h] * IS3;
  x[i] += g3;
  int b = batch[n];
  float dv[3]; float d2 = 0.0f;
#pragma unroll
  for (int d = 0; d < 3; ++d) {
    dv[d] = cell[b * 9 + k * 3 + d] - pos[n * 3 + d];
    d2 += dv[d] * dv[d];
  }
  float inv = -1.0f / sqrtf(d2);
#pragma unroll
  for (int d = 0; d < 3; ++d) {
    int vi = (n * 3 + d) * HD + h;
    vec[vi] = (g1 * vec[vi] + g2 * dv[d] * inv) * ISH;
  }
}
__global__ void k_cat_scal(const float* __restrict__ x, const float* __restrict__ scal,
                           const int* __restrict__ batch, float* __restrict__ cat) {
  int i = blockIdx.x * blockDim.x + threadIdx.x;
  if (i >= NNODE * HD) return;
  int n = i / HD, h = i % HD;
  cat[n * 256 + h]      = x[i];
  cat[n * 256 + 128 + h] = scal[batch[n] * HD + h];
}
__global__ void k_addvecl(float* __restrict__ out, const float* __restrict__ vec,
                          const float* __restrict__ vecl, const int* __restrict__ batch) {
  int i = blockIdx.x * blockDim.x + threadIdx.x;
  if (i >= NNODE * 3 * HD) return;
  int n = i / (3 * HD), rem = i % (3 * HD);
  out[i] = vec[i] + vecl[batch[n] * 3 * HD + rem];
}
__global__ void k_mp_post(float* __restrict__ x, const float* __restrict__ dx,
                          float* __restrict__ vec, const float* __restrict__ dvec) {
  int i = blockIdx.x * blockDim.x + threadIdx.x;
  if (i >= NNODE * 3 * HD) return;
  vec[i] += dvec[i];
  if (i < NNODE * HD) x[i] = (x[i] + dx[i]) * IS2;
}
__global__ void k_norm_cat(const float* __restrict__ first, const float* __restrict__ v12,
                           float* __restrict__ cat, int n) {
  int i = blockIdx.x * blockDim.x + threadIdx.x;
  if (i >= n * HD) return;
  int r = i / HD, h = i % HD;
  float s = 0.0f;
#pragma unroll
  for (int d = 0; d < 3; ++d) {
    float v = v12[(r * 3 + d) * 256 + 128 + h];
    s += v * v;
  }
  cat[r * 256 + h]       = first[i];
  cat[r * 256 + 128 + h] = sqrtf(s + 1e-8f);
}
__global__ void k_su_apply(float* __restrict__ x, float* __restrict__ vec,
                           const float* __restrict__ xh, const float* __restrict__ v12) {
  int i = blockIdx.x * blockDim.x + threadIdx.x;
  if (i >= NNODE * HD) return;
  int n = i / HD, h = i % HD;
  float x1 = xh[n * 384 + h];
  float x2 = xh[n * 384 + 128 + h];
  float x3 = xh[n * 384 + 256 + h];
  float ox = x[i];
  x[i] = ox + x2 * IS2 + ox * tanhf(x3);
#pragma unroll
  for (int d = 0; d < 3; ++d) {
    int vi = (n * 3 + d) * HD + h;
    vec[vi] += x1 * v12[(n * 3 + d) * 256 + h];
  }
}
__global__ void k_struct(const float* __restrict__ att, const float* __restrict__ vec,
                         const float* __restrict__ wp, float* __restrict__ pos) {
  __shared__ float red[128];
  int n = blockIdx.x, h = threadIdx.x;
  float a = att[n * HD + h];
  red[h] = a; __syncthreads();
  for (int s = 64; s > 0; s >>= 1) { if (h < s) red[h] = fmaxf(red[h], red[h + s]); __syncthreads(); }
  float mx = red[0]; __syncthreads();
  float e = __expf(a - mx);
  red[h] = e; __syncthreads();
  for (int s = 64; s > 0; s >>= 1) { if (h < s) red[h] += red[h + s]; __syncthreads(); }
  float p = e / red[0]; __syncthreads();
  float w = wp[h] + p;
  for (int d = 0; d < 3; ++d) {
    red[h] = w * vec[(n * 3 + d) * HD + h]; __syncthreads();
    for (int s = 64; s > 0; s >>= 1) { if (h < s) red[h] += red[h + s]; __syncthreads(); }
    if (h == 0) pos[n * 3 + d] += red[0] * 0.5f;
    __syncthreads();
  }
}
__global__ void k_seg_accum(const float* __restrict__ x, const float* __restrict__ vec,
                            const int* __restrict__ batch, float* __restrict__ mx,
                            float* __restrict__ mv) {
  int i = blockIdx.x * blockDim.x + threadIdx.x;
  if (i >= NNODE * 3 * HD) return;
  int n = i / (3 * HD), rem = i % (3 * HD);
  int b = batch[n];
  atomAddF(&mv[b * 3 * HD + rem], vec[i]);
  if (rem < HD) atomAddF(&mx[b * HD + rem], x[n * HD + rem]);
}
__global__ void k_gcat(const float* __restrict__ mx, const float* __restrict__ scal,
                       float* __restrict__ cat) {
  int i = blockIdx.x * blockDim.x + threadIdx.x;
  if (i >= NGR * HD) return;
  int b = i / HD, h = i % HD;
  cat[b * 256 + h]       = mx[i] * (1.0f / (float)NPG);
  cat[b * 256 + 128 + h] = scal[i];
}
__global__ void k_gmv(const float* __restrict__ mv, const float* __restrict__ vecl,
                      float* __restrict__ gt) {
  int i = blockIdx.x * blockDim.x + threadIdx.x;
  if (i < NGR * 3 * HD) gt[i] = mv[i] * (1.0f / (float)NPG) + vecl[i];
}
__global__ void k_gupd1(float* __restrict__ scal, const float* __restrict__ gst,
                        float* __restrict__ vecl, const float* __restrict__ gvt) {
  int i = blockIdx.x * blockDim.x + threadIdx.x;
  if (i >= NGR * 3 * HD) return;
  vecl[i] += gvt[i];
  if (i < NGR * HD) scal[i] += gst[i];
}
__global__ void k_gupd2(float* __restrict__ scal, float* __restrict__ vecl,
                        const float* __restrict__ gs3, const float* __restrict__ gv12) {
  int i = blockIdx.x * blockDim.x + threadIdx.x;
  if (i >= NGR * HD) return;
  int b = i / HD, h = i % HD;
  float s1 = gs3[b * 384 + h];
  float s2 = gs3[b * 384 + 128 + h];
  float s3 = gs3[b * 384 + 256 + h];
  float sl = scal[i];
  scal[i] = s2 + sl * tanhf(s3);
#pragma unroll
  for (int d = 0; d < 3; ++d) {
    int vi = (b * 3 + d) * HD + h;
    vecl[vi] += s1 * gv12[(b * 3 + d) * 256 + h];
  }
}
__global__ void k_celldelta(const float* __restrict__ vecl, const float* __restrict__ wl,
                            float* __restrict__ cell, int k) {
  int t = blockIdx.x * blockDim.x + threadIdx.x;
  if (t >= NGR * 3) return;
  int b = t / 3, d = t % 3;
  float s = 0.0f;
  for (int h = 0; h < HD; ++h) s += vecl[(b * 3 + d) * HD + h] * wl[h];
  cell[b * 9 + k * 3 + d] += s;
}

// ---------------- host orchestration ----------------

struct LatP {
  const float *elp_w, *elp_b, *ml_w;
  const float *msg0_w, *msg0_b, *msg1_w, *msg1_b;
  const float *msl0_w, *msl0_b, *msl1_w, *msl1_b;
  const float *mvg_w, *mvl_w;
  const float *slp0_w, *slp0_b, *slp1_w, *slp1_b;
  const float *vlp_w;
  const float *xlp0_w, *xlp0_b, *xlp1_w, *xlp1_b;
};
struct LayerP {
  LatP lat[3];
  const float *ep_w, *ep_b, *xp0_w, *xp0_b, *xp1_w, *xp1_b;
  const float *ap0_w, *ap0_b, *ap1_w, *ap1_b, *pm_w;
  const float *vp_w, *xv0_w, *xv0_b, *xv1_w, *xv1_b;
};

#define EWL(kern, n, ...) kern<<<((n) + 255) / 256, 256, 0, stream>>>(__VA_ARGS__)

extern "C" void kernel_launch(void* const* d_in, const int* in_sizes, int n_in,
                              void* d_out, int out_size, void* d_ws, size_t ws_size,
                              hipStream_t stream) {
  (void)in_sizes; (void)n_in; (void)out_size; (void)ws_size;
  const float* pos0  = (const float*)d_in[0];
  const float* cell0 = (const float*)d_in[1];
  const float* cof   = (const float*)d_in[2];

  int cur = 3;
  auto nxt  = [&]() -> const float* { return (const float*)d_in[cur++]; };
  auto lin  = [&](const float** w, const float** b) { *b = nxt(); *w = nxt(); };  // b then w
  auto linw = [&](const float** w) { *w = nxt(); };

  const float* atom_emb = nxt();
  LayerP L[3];
  for (int l = 0; l < 3; ++l) {
    for (int k = 0; k < 3; ++k) {
      LatP& q = L[l].lat[k];
      lin(&q.elp_w, &q.elp_b);
      linw(&q.ml_w);
      lin(&q.msg0_w, &q.msg0_b); lin(&q.msg1_w, &q.msg1_b);
      lin(&q.msl0_w, &q.msl0_b); lin(&q.msl1_w, &q.msl1_b);
      linw(&q.mvg_w); linw(&q.mvl_w);
      lin(&q.slp0_w, &q.slp0_b); lin(&q.slp1_w, &q.slp1_b);
      linw(&q.vlp_w);
      lin(&q.xlp0_w, &q.xlp0_b); lin(&q.xlp1_w, &q.xlp1_b);
    }
    lin(&L[l].ep_w, &L[l].ep_b);
    lin(&L[l].xp0_w, &L[l].xp0_b); lin(&L[l].xp1_w, &L[l].xp1_b);
    lin(&L[l].ap0_w, &L[l].ap0_b); lin(&L[l].ap1_w, &L[l].ap1_b);
    linw(&L[l].pm_w);
    linw(&L[l].vp_w);
    lin(&L[l].xv0_w, &L[l].xv0_b); lin(&L[l].xv1_w, &L[l].xv1_b);
  }
  const float* scalar_emb = nxt();
  const int* eidx   = (const int*)d_in[cur++];
  cur++;  // neighbors (uniform, hardcoded EPG)
  const int* batchp = (const int*)d_in[cur++];
  const int* zp     = (const int*)d_in[cur++];
  const int* jidx = eidx, * iidx = eidx + NEDGE;

  // workspace carve-up (all slices stay 16B aligned)
  char* wp = (char*)d_ws;
  auto A = [&](size_t nf) { float* r = (float*)wp; wp += nf * sizeof(float); return r; };
  float* xbuf  = A((size_t)NNODE * HD);
  float* vecb  = A((size_t)NNODE * 3 * HD);
  float* pcur  = A((size_t)NNODE * 3);
  float* ccur  = A(72);
  float* hbuf  = A((size_t)NNODE * HD);
  float* xp    = A((size_t)NNODE * 3 * HD);
  float* t3h   = A((size_t)NNODE * 3 * HD);
  float* catb  = A((size_t)NNODE * 2 * HD);
  float* rbfn  = A((size_t)NNODE * NRBF);
  float* dxb   = A((size_t)NNODE * HD);
  float* dvecb = A((size_t)NNODE * 3 * HD);
  float* v12   = A((size_t)NNODE * 3 * 2 * HD);
  float* attb  = A((size_t)NNODE * HD);
  float* scal  = A(3 * NGR * HD);
  float* vecl  = A(3 * NGR * 3 * HD);
  float* mx    = A(NGR * HD);
  float* mv    = A(NGR * 3 * HD);
  float* gcat  = A(NGR * 2 * HD);
  float* gh    = A(NGR * HD);
  float* gst   = A(NGR * HD);
  float* gt    = A(NGR * 3 * HD);
  float* gvt   = A(NGR * 3 * HD);
  float* gv12  = A(NGR * 3 * 2 * HD);
  float* gs3   = A(NGR * 3 * HD);
  // f16 weight arena (bump; deterministic sequence each call)
  auto AH = [&](size_t nh) {
    _Float16* r = (_Float16*)wp; wp += ((nh + 7) & ~(size_t)7) * sizeof(_Float16); return r;
  };

  // GEMM with on-the-fly weight transpose/convert to f16.
  auto gemm = [&](const float* X, const float* W, const float* B, float* Y,
                  int M, int N, int K, int act) {
    _Float16* wt = AH((size_t)K * N);
    EWL(k_cvtT, K * N, W, wt, K, N);
    dim3 g((N + 127) / 128, (M + 31) / 32);
    k_gemm<<<g, 256, 0, stream>>>(X, wt, B, Y, M, N, K, act);
  };

  // init
  EWL(k_embed, NNODE * HD, atom_emb, zp, xbuf);
  EWL(k_zero, NNODE * 3 * HD, vecb, NNODE * 3 * HD);
  EWL(k_copy, NNODE * 3, pos0, pcur, NNODE * 3);
  EWL(k_copy, 72, cell0, ccur, 72);
  EWL(k_init_scal, 3 * NGR * HD, scalar_emb, scal);
  EWL(k_zero, 3 * NGR * 3 * HD, vecl, 3 * NGR * 3 * HD);

  for (int l = 0; l < 3; ++l) {
    const LayerP& P = L[l];
    // ---- lattice local x3 ----
    for (int k = 0; k < 3; ++k) {
      const LatP& q = P.lat[k];
      float* sk = scal + k * NGR * HD;
      float* vk = vecl + k * NGR * 3 * HD;
      EWL(k_node_rbf, NNODE * NRBF, pcur, ccur, batchp, k, rbfn);
      gemm(rbfn, q.elp_w, q.elp_b, t3h, NNODE, 384, 128, 0);           // e_p
      gemm(xbuf, q.xlp0_w, q.xlp0_b, hbuf, NNODE, 128, 128, 1);
      gemm(hbuf, q.xlp1_w, q.xlp1_b, xp, NNODE, 384, 128, 0);          // x_p
      EWL(k_latloc_apply, NNODE * HD, xbuf, vecb, t3h, xp, pcur, ccur, batchp, k);
      EWL(k_cat_scal, NNODE * HD, xbuf, sk, batchp, catb);
      gemm(catb, q.msl0_w, q.msl0_b, hbuf, NNODE, 128, 256, 1);
      gemm(hbuf, q.msl1_w, q.msl1_b, t3h, NNODE, 128, 128, 1);         // act_last
      EWL(k_add, NNODE * HD, xbuf, t3h, NNODE * HD);
      EWL(k_addvecl, NNODE * 3 * HD, t3h, vecb, vk, batchp);
      gemm(t3h, q.mvl_w, nullptr, xp, 3 * NNODE, 128, 128, 0);
      EWL(k_add, NNODE * 3 * HD, vecb, xp, NNODE * 3 * HD);
    }
    // ---- message passing (fused RBF + WMMA + scatter) ----
    gemm(xbuf, P.xp0_w, P.xp0_b, hbuf, NNODE, 64, 128, 1);
    gemm(hbuf, P.xp1_w, P.xp1_b, xp, NNODE, 384, 64, 0);               // x_h
    {
      _Float16* epT = AH((size_t)128 * 384);
      EWL(k_cvtT, 128 * 384, P.ep_w, epT, 128, 384);
      EWL(k_zero, NNODE * HD, dxb, NNODE * HD);
      EWL(k_zero, NNODE * 3 * HD, dvecb, NNODE * 3 * HD);
      k_edge_mp<<<NEDGE / 16, 256, 0, stream>>>(pcur, ccur, cof, jidx, iidx,
                                                xp, vecb, epT, P.ep_b, dxb, dvecb);
    }
    EWL(k_mp_post, NNODE * 3 * HD, xbuf, dxb, vecb, dvecb);
    // ---- self update ----
    gemm(vecb, P.vp_w, nullptr, v12, 3 * NNODE, 256, 128, 0);
    EWL(k_norm_cat, NNODE * HD, xbuf, v12, catb, NNODE);
    gemm(catb, P.xv0_w, P.xv0_b, hbuf, NNODE, 128, 256, 1);
    gemm(hbuf, P.xv1_w, P.xv1_b, t3h, NNODE, 384, 128, 0);
    EWL(k_su_apply, NNODE * HD, xbuf, vecb, t3h, v12);
    // ---- structure update ----
    gemm(xbuf, P.ap0_w, P.ap0_b, hbuf, NNODE, 128, 128, 1);
    gemm(hbuf, P.ap1_w, P.ap1_b, attb, NNODE, 128, 128, 0);
    k_struct<<<NNODE, 128, 0, stream>>>(attb, vecb, P.pm_w, pcur);
    // ---- lattice global x3 ----
    EWL(k_zero, NGR * HD, mx, NGR * HD);
    EWL(k_zero, NGR * 3 * HD, mv, NGR * 3 * HD);
    EWL(k_seg_accum, NNODE * 3 * HD, xbuf, vecb, batchp, mx, mv);
    for (int k = 0; k < 3; ++k) {
      const LatP& q = P.lat[k];
      float* sk = scal + k * NGR * HD;
      float* vk = vecl + k * NGR * 3 * HD;
      EWL(k_gcat, NGR * HD, mx, sk, gcat);
      gemm(gcat, q.msg0_w, q.msg0_b, gh, NGR, 128, 256, 1);
      gemm(gh, q.msg1_w, q.msg1_b, gst, NGR, 128, 128, 1);             // act_last
      EWL(k_gmv, NGR * 3 * HD, mv, vk, gt);
      gemm(gt, q.mvg_w, nullptr, gvt, 3 * NGR, 128, 128, 0);
      EWL(k_gupd1, NGR * 3 * HD, sk, gst, vk, gvt);
      gemm(vk, q.vlp_w, nullptr, gv12, 3 * NGR, 256, 128, 0);
      EWL(k_norm_cat, NGR * HD, sk, gv12, gcat, NGR);
      gemm(gcat, q.slp0_w, q.slp0_b, gh, NGR, 128, 256, 1);
      gemm(gh, q.slp1_w, q.slp1_b, gs3, NGR, 384, 128, 0);
      EWL(k_gupd2, NGR * HD, sk, vk, gs3, gv12);
      k_celldelta<<<1, 32, 0, stream>>>(vk, q.ml_w, ccur, k);
    }
  }

  float* out = (float*)d_out;
  EWL(k_copy, NNODE * 3, pcur, out, NNODE * 3);
  EWL(k_copy, 72, ccur, out + NNODE * 3, 72);
}
